// LinearGaussianVQVAE_62904091017905
// MI455X (gfx1250) — matrix operand order repaired and checked by reading, not compiled, not run on hardware
//
#include <hip/hip_runtime.h>

// ---------------------------------------------------------------------------
// Linear Gaussian VQ-VAE forward on gfx1250 (CDNA5), wave32 + WMMA f16 +
// async global<->LDS staging.
//   B=131072, D=512, K=64, CB=1024
// d_in: x (B*D f32), U_k (D*K f32), codebook (CB*K f32)
// d_out: [x_recon B*D | z B*K | z_q B*K | indices B]  (float)
// ---------------------------------------------------------------------------

#define Bn   131072
#define Dn   512
#define Kn   64
#define CBn  1024

typedef __attribute__((ext_vector_type(16))) _Float16 v16h;
typedef __attribute__((ext_vector_type(8)))  float    v8f;
typedef __attribute__((ext_vector_type(4)))  float    f4;

__device__ __forceinline__ v8f wmma_f16(v16h a, v16h b, v8f c) {
  return __builtin_amdgcn_wmma_f32_16x16x32_f16(false, a, false, b, (short)0,
                                                c, false, false);
}

// Async DMA: global -> LDS, 16 bytes per lane (ASYNCcnt tracked).
__device__ __forceinline__ void async_ld16(void* lds_dst, const void* gsrc) {
  unsigned ldsa = (unsigned)(unsigned long long)lds_dst;  // low 32b = LDS offset
  unsigned long long ga = (unsigned long long)gsrc;
  asm volatile("global_load_async_to_lds_b128 %0, %1, off"
               :: "v"(ldsa), "v"(ga) : "memory");
}

// Async DMA: LDS -> global, 16 bytes per lane.
__device__ __forceinline__ void async_st16(void* gdst, const void* lds_src) {
  unsigned ldsa = (unsigned)(unsigned long long)lds_src;
  unsigned long long ga = (unsigned long long)gdst;
  asm volatile("global_store_async_from_lds_b128 %1, %0, off"
               :: "v"(ldsa), "v"(ga) : "memory");
}

// Build a 16x32 f16 A-fragment for one lane from 32 consecutive-K floats of a
// row.  CDNA5 16-bit A layout: lanes 0-15 hold K runs [+0..+7],[+16..+23];
// lanes 16-31 hold [+8..+15],[+24..+31].  Caller passes base already offset by
// (+0 or +8); we read runs at +0 and +16 floats relative to base.
__device__ __forceinline__ v16h make_a_frag(const float* base) {
  const f4* q = (const f4*)base;
  f4 q0 = q[0], q1 = q[1], q2 = q[4], q3 = q[5];
  v16h a;
  a[0]  = (_Float16)q0[0]; a[1]  = (_Float16)q0[1];
  a[2]  = (_Float16)q0[2]; a[3]  = (_Float16)q0[3];
  a[4]  = (_Float16)q1[0]; a[5]  = (_Float16)q1[1];
  a[6]  = (_Float16)q1[2]; a[7]  = (_Float16)q1[3];
  a[8]  = (_Float16)q2[0]; a[9]  = (_Float16)q2[1];
  a[10] = (_Float16)q2[2]; a[11] = (_Float16)q2[3];
  a[12] = (_Float16)q3[0]; a[13] = (_Float16)q3[1];
  a[14] = (_Float16)q3[2]; a[15] = (_Float16)q3[3];
  return a;
}

// ---------------------------------------------------------------------------
// Prep 1: pack U_k and codebook^T into WMMA B-fragment order (f16); 0.5*||c||^2.
// B-fragment (32x16 f16): lanes 0-15 K=[0..7],[16..23], lanes 16-31
// K=[8..15],[24..31], column n = lane%16.
// ---------------------------------------------------------------------------
__global__ __launch_bounds__(256) void vqvae_prep_pack(
    const float* __restrict__ U, const float* __restrict__ C,
    _Float16* __restrict__ u_pk, _Float16* __restrict__ cb_pk,
    float* __restrict__ csq) {
  const int t = threadIdx.x;

  for (int s = t; s < 2048; s += 256) {          // U_k: 64 frags * 32 lanes
    const int f = s >> 5, l = s & 31;
    const int ks = f >> 2, nt = f & 3;
    const int kb = ks * 32 + ((l & 16) ? 8 : 0);
    const int n  = nt * 16 + (l & 15);
#pragma unroll
    for (int i = 0; i < 16; ++i)
      u_pk[s * 16 + i] = (_Float16)U[(kb + i + (i & 8)) * Kn + n];
  }

  for (int s = t; s < 4096; s += 256) {          // cb^T: 128 frags * 32 lanes
    const int g = s >> 5, l = s & 31;
    const int ct = g >> 1, k2 = g & 1;
    const int kb  = k2 * 32 + ((l & 16) ? 8 : 0);
    const int col = ct * 16 + (l & 15);
#pragma unroll
    for (int i = 0; i < 16; ++i)
      cb_pk[s * 16 + i] = (_Float16)C[col * Kn + kb + i + (i & 8)];
  }

  for (int c = t; c < CBn; c += 256) {           // 0.5*||c||^2
    float acc = 0.f;
#pragma unroll 8
    for (int k = 0; k < Kn; ++k) { float v = C[c * Kn + k]; acc += v * v; }
    csq[c] = 0.5f * acc;
  }
}

// ---------------------------------------------------------------------------
// Prep 2: decoded codebook  Cdec[c][d] = sum_k codebook[c][k] * U_k[d][k]
// (2 MB, L2-resident; turns the decode GEMM into a row gather)
// ---------------------------------------------------------------------------
__global__ __launch_bounds__(256) void vqvae_prep_cdec(
    const float* __restrict__ U, const float* __restrict__ C,
    float* __restrict__ cdec) {
  const int tid = blockIdx.x * 256 + threadIdx.x;
  const int c = tid >> 9, d = tid & 511;
  float acc = 0.f;
#pragma unroll 8
  for (int k = 0; k < Kn; ++k) acc += C[c * Kn + k] * U[d * Kn + k];
  cdec[tid] = acc;
}

// ---------------------------------------------------------------------------
// Main: 512 blocks * 256 threads (8 waves); each wave owns 32 rows (M=32:
// one B fragment feeds two A fragments -> 2x arithmetic intensity on LDS).
// LDS: u_pk 64K + cb_pk 128K + csq 4K + z 64K + idx 1K = 261 KB (<320 KB/WGP).
// ---------------------------------------------------------------------------
__global__ __launch_bounds__(256) void vqvae_main(
    const float* __restrict__ x, const float* __restrict__ codebook,
    const _Float16* __restrict__ u_pk, const _Float16* __restrict__ cb_pk,
    const float* __restrict__ csq, const float* __restrict__ cdec,
    float* __restrict__ out_xr, float* __restrict__ out_z,
    float* __restrict__ out_zq, float* __restrict__ out_idx) {
  __shared__ __align__(16) _Float16 u_lds[32768];    // 64 KB
  __shared__ __align__(16) _Float16 cb_lds[65536];   // 128 KB
  __shared__ __align__(16) float    csq_lds[CBn];    // 4 KB
  __shared__ __align__(16) float    z_lds[8 * 32 * Kn];  // 64 KB (8 KB/wave)
  __shared__ int idx_lds[8 * 32];                    // 1 KB

  const int t    = threadIdx.x;
  const int wave = t >> 5;
  const int lane = t & 31;
  const int n    = lane & 15;
  const int h    = lane >> 4;

  // ---- Async-DMA the packed operands into LDS (no VGPR round-trip) ----
  for (int s = t; s < 4096; s += 256)              // u_pk: 64 KB
    async_ld16((char*)u_lds + s * 16, (const char*)u_pk + s * 16);
  for (int s = t; s < 8192; s += 256)              // cb_pk: 128 KB
    async_ld16((char*)cb_lds + s * 16, (const char*)cb_pk + s * 16);
  async_ld16((char*)csq_lds + t * 16, (const char*)csq + t * 16);  // 4 KB
  asm volatile("s_wait_asynccnt 0x0" ::: "memory");
  __syncthreads();

  const int row0 = blockIdx.x * 256 + wave * 32;
  float* zl = z_lds + wave * 32 * Kn;

  // ---- Encode: z(32x64) = x_tile(32x512) @ U(512x64) ----
  v8f zacc0[4] = {}, zacc1[4] = {};
#pragma unroll 2
  for (int ks = 0; ks < 16; ++ks) {
    const float* xb0 = x + (size_t)(row0 + n) * Dn + ks * 32 + (h ? 8 : 0);
    v16h a0 = make_a_frag(xb0);
    v16h a1 = make_a_frag(xb0 + 16 * Dn);
#pragma unroll
    for (int nt = 0; nt < 4; ++nt) {
      v16h b = *(const v16h*)(u_lds + ((ks * 4 + nt) * 32 + lane) * 16);
      zacc0[nt] = wmma_f16(a0, b, zacc0[nt]);
      zacc1[nt] = wmma_f16(a1, b, zacc1[nt]);
    }
  }

  // Stage z tiles to LDS (C/D layout: M = r + 8*h, N = nt*16 + n).
#pragma unroll
  for (int nt = 0; nt < 4; ++nt)
#pragma unroll
    for (int r = 0; r < 8; ++r) {
      zl[(r + 8 * h) * Kn + nt * 16 + n]        = zacc0[nt][r];
      zl[(16 + r + 8 * h) * Kn + nt * 16 + n]   = zacc1[nt][r];
    }
  asm volatile("s_wait_dscnt 0x0" ::: "memory");   // wave-local LDS RAW

  // z out: async DMA straight from the LDS staging tile (8 KB contiguous).
  {
    char* gdst = (char*)(out_z + (size_t)row0 * Kn);
    for (int s = lane; s < 512; s += 32)
      async_st16(gdst + s * 16, (const char*)zl + s * 16);
  }

  // ---- Score: minimize 0.5*||c||^2 - z.c over 1024 codes ----
  v16h za0[2], za1[2];
#pragma unroll
  for (int s = 0; s < 2; ++s) {
    za0[s] = make_a_frag(zl + n * Kn + s * 32 + (h ? 8 : 0));
    za1[s] = make_a_frag(zl + (16 + n) * Kn + s * 32 + (h ? 8 : 0));
  }

  float bv0[8], bv1[8];
  int   bc0[8], bc1[8];
#pragma unroll
  for (int r = 0; r < 8; ++r) {
    bv0[r] = 3.4e38f; bv1[r] = 3.4e38f; bc0[r] = 0; bc1[r] = 0;
  }

#pragma unroll 2
  for (int ct = 0; ct < 64; ++ct) {
    v8f acc0 = {}, acc1 = {};
#pragma unroll
    for (int s = 0; s < 2; ++s) {
      v16h b = *(const v16h*)(cb_lds + ((ct * 2 + s) * 32 + lane) * 16);
      acc0 = wmma_f16(za0[s], b, acc0);
      acc1 = wmma_f16(za1[s], b, acc1);
    }
    const float bias = csq_lds[ct * 16 + n];
#pragma unroll
    for (int r = 0; r < 8; ++r) {
      float v0 = bias - acc0[r];
      bool l0 = v0 < bv0[r];
      bv0[r] = l0 ? v0 : bv0[r];
      bc0[r] = l0 ? ct : bc0[r];
      float v1 = bias - acc1[r];
      bool l1 = v1 < bv1[r];
      bv1[r] = l1 ? v1 : bv1[r];
      bc1[r] = l1 ? ct : bc1[r];
    }
  }

  // Cross-lane argmin over the 16 columns per row (XOR butterfly, wave32).
#pragma unroll
  for (int r = 0; r < 8; ++r) {
    float v = bv0[r];
    int   c = bc0[r] * 16 + n;
#pragma unroll
    for (int off = 1; off < 16; off <<= 1) {
      const float ov = __shfl_xor(v, off, 32);
      const int   oc = __shfl_xor(c, off, 32);
      if (ov < v) { v = ov; c = oc; }
    }
    if (n == 0) idx_lds[wave * 32 + r + 8 * h] = c;

    float w = bv1[r];
    int   d = bc1[r] * 16 + n;
#pragma unroll
    for (int off = 1; off < 16; off <<= 1) {
      const float ow = __shfl_xor(w, off, 32);
      const int   od = __shfl_xor(d, off, 32);
      if (ow < w) { w = ow; d = od; }
    }
    if (n == 0) idx_lds[wave * 32 + 16 + r + 8 * h] = d;
  }
  asm volatile("s_wait_dscnt 0x0" ::: "memory");

  // Indices (as float, exact for idx < 1024): 32 lanes cover the 32 rows.
  out_idx[row0 + lane] = (float)idx_lds[wave * 32 + lane];

  // z_q gather (codebook rows) and x_recon gather (L2-hot Cdec rows).
#pragma unroll 1
  for (int r = 0; r < 32; ++r) {
    const int c = idx_lds[wave * 32 + r];
    out_zq[(size_t)(row0 + r) * Kn + lane]      = codebook[c * Kn + lane];
    out_zq[(size_t)(row0 + r) * Kn + 32 + lane] = codebook[c * Kn + 32 + lane];
    const f4* src = (const f4*)(cdec + (size_t)c * Dn);
    f4* dst = (f4*)(out_xr + (size_t)(row0 + r) * Dn);
#pragma unroll
    for (int j = 0; j < 4; ++j) dst[j * 32 + lane] = src[j * 32 + lane];
  }
}

// ---------------------------------------------------------------------------
extern "C" void kernel_launch(void* const* d_in, const int* in_sizes, int n_in,
                              void* d_out, int out_size, void* d_ws,
                              size_t ws_size, hipStream_t stream) {
  const float* x = (const float*)d_in[0];
  const float* U = (const float*)d_in[1];
  const float* C = (const float*)d_in[2];

  // Workspace layout (~2.3 MB total)
  char* ws = (char*)d_ws;
  float*    cdec  = (float*)ws;                        // 1024*512*4 = 2 MB
  _Float16* u_pk  = (_Float16*)(ws + 2097152);         // 64 KB
  _Float16* cb_pk = (_Float16*)(ws + 2162688);         // 128 KB
  float*    csq   = (float*)(ws + 2293760);            // 4 KB

  float* out     = (float*)d_out;
  float* out_xr  = out;
  float* out_z   = out + (size_t)Bn * Dn;
  float* out_zq  = out_z + (size_t)Bn * Kn;
  float* out_idx = out_zq + (size_t)Bn * Kn;

  vqvae_prep_pack<<<1, 256, 0, stream>>>(U, C, u_pk, cb_pk, csq);
  vqvae_prep_cdec<<<(CBn * Dn) / 256, 256, 0, stream>>>(U, C, cdec);
  vqvae_main<<<Bn / 256, 256, 0, stream>>>(x, C, u_pk, cb_pk, csq, cdec,
                                           out_xr, out_z, out_zq, out_idx);
}
